// AttentionMemory_6219112644708
// MI455X (gfx1250) — matrix operand order repaired and checked by reading, not compiled, not run on hardware
//
#include <hip/hip_runtime.h>

// CDNA5 (gfx1250) wave32 WMMA types
typedef __attribute__((ext_vector_type(16))) __bf16 v16bf;
typedef __attribute__((ext_vector_type(8)))  float  v8f;
typedef __attribute__((ext_vector_type(4)))  unsigned int v4u;

namespace {
constexpr int kB     = 4;
constexpr int kCK    = 64;
constexpr int kN     = 4032;       // memory axis (softmax axis), H*W
constexpr int kM     = 4032;       // query axis, H*W
constexpr int kNT    = kN / 16;    // 252 row tiles (== column tiles, kN == kM)
constexpr int kWAVES = 8;          // 256 threads = 8 wave32

// exp(z) = exp2(z*log2e); z = (2*dot - a - c)/8  ->  fold /8 and log2e together
constexpr float kScale  = 0.18033688011112042f;   // log2(e)/8
constexpr float kScale2 = 0.36067376022224083f;   // 2*log2(e)/8

// workspace layout (bytes)
constexpr size_t kStageBytes = (size_t)kB * kNT * 32 * 32 * 2; // bf16 A fragments: 2,064,384
constexpr size_t kAnormOff   = kStageBytes;                    // kB*kN f32
constexpr size_t kCnormOff   = kAnormOff + (size_t)kB * kN * 4;
constexpr size_t kWsNeeded   = kCnormOff + (size_t)kB * kM * 4;
}

__device__ __forceinline__ __bf16 f2bf(float f) { return (__bf16)f; }

// A-fragment K index, 16-bit A 16x32 (ISA 7.12.2): lanes 0-15 cover K 0..7,16..23
__device__ __forceinline__ int ka_idx(int e, int hlf) {
    return (e < 8) ? (e + 8 * hlf) : (e + 8 + 8 * hlf);
}
// B-fragment K index, 16-bit B 32x16: same K order as A (lanes 0-15 -> K 0..7,16..23)
__device__ __forceinline__ int kb_idx(int e, int hlf) {
    return (e < 8) ? (e + 8 * hlf) : (e + 8 + 8 * hlf);
}

// ---------------------------------------------------------------------------
// Pre-pass: one wave per (tile, batch).
//   * converts mk to bf16 laid out exactly as per-lane WMMA A fragments
//     (64-byte slot per lane -> main loop uses 4x global_load_b128 per tile)
//   * computes row norms ||mk_n||^2 (lanes 0-15) and col norms ||qk_m||^2 (lanes 16-31)
// ---------------------------------------------------------------------------
__global__ __launch_bounds__(32)
void prepass_kernel(const float* __restrict__ mk, const float* __restrict__ qk,
                    __bf16* __restrict__ stage, float* __restrict__ anorm,
                    float* __restrict__ cnorm)
{
    const int t    = blockIdx.x;
    const int b    = blockIdx.y;
    const int lane = threadIdx.x;
    const int hlf  = lane >> 4;
    const int lcol = lane & 15;
    const int n0   = t * 16;

    const float* mkb = mk + (size_t)b * kCK * kN;
    const float* qkb = qk + (size_t)b * kCK * kM;

    // bf16 A-fragment staging: slot = ((b*kNT + t)*32 + lane) * 32 elements
    __bf16* s = stage + ((size_t)(b * kNT + t) * 32 + lane) * 32;
    const float* p = mkb + n0 + lcol;
    #pragma unroll
    for (int e = 0; e < 16; ++e) {
        int k = ka_idx(e, hlf);
        s[e]      = f2bf(p[(size_t)k * kN]);        // a0 (K 0..31 block)
        s[16 + e] = f2bf(p[(size_t)(k + 32) * kN]); // a1 (K 32..63 block)
    }

    // norms (exact f32), computed once per (b, index)
    if (hlf == 0) {
        float ssum = 0.f;
        const float* r = mkb + n0 + lcol;
        #pragma unroll 8
        for (int k = 0; k < kCK; ++k) { float v = r[(size_t)k * kN]; ssum += v * v; }
        anorm[b * kN + n0 + lcol] = ssum;
    } else {
        float ssum = 0.f;
        const float* r = qkb + n0 + lcol;
        #pragma unroll 8
        for (int k = 0; k < kCK; ++k) { float v = r[(size_t)k * kM]; ssum += v * v; }
        cnorm[b * kM + n0 + lcol] = ssum;
    }
}

// ---------------------------------------------------------------------------
// Main kernel: one WG per (16-wide column strip, batch). Two passes over the
// 4032-row strip: (1) sum of exp(affinity) per column, (2) normalize + stream.
// ---------------------------------------------------------------------------
__global__ __launch_bounds__(256)
void affinity_softmax_fast(const float* __restrict__ qk,
                           const __bf16* __restrict__ stage,
                           const float* __restrict__ anorm_g,
                           const float* __restrict__ cnorm_g,
                           float* __restrict__ out)
{
    const int b    = blockIdx.y;
    const int m0   = blockIdx.x * 16;
    const int tid  = threadIdx.x;
    const int lane = tid & 31;
    const int wave = tid >> 5;
    const int hlf  = lane >> 4;
    const int lcol = lane & 15;

    __shared__ float anorm[kN];                 // 16 KB, pre-scaled by kScale
    __shared__ float redbuf[kWAVES][16];
    __shared__ float colrecip[16];

    // stream precomputed row norms into LDS, pre-scaled so the hot loop needs
    // only pk_fma + pk_add + v_exp per element pair
    {
        const float* ag = anorm_g + b * kN;
        for (int n = tid; n < kN; n += 256) anorm[n] = ag[n] * kScale;
    }
    const float cc = cnorm_g[b * kM + m0 + lcol] * kScale;

    // B fragments: invariant across all row tiles, pinned in VGPRs
    v16bf bf0, bf1;
    {
        const float* q = qk + (size_t)b * kCK * kM + m0 + lcol;
        #pragma unroll
        for (int e = 0; e < 16; ++e) {
            int k = kb_idx(e, hlf);
            bf0[e] = f2bf(q[(size_t)k * kM]);
            bf1[e] = f2bf(q[(size_t)(k + 32) * kM]);
        }
    }
    __syncthreads();

    const __bf16* sbase = stage + (size_t)(b * kNT) * 32 * 32 + (size_t)lane * 32;

    // ================= PASS 1: per-column sum of exp(affinity) ==============
    float psum0 = 0.f, psum1 = 0.f;
    for (int t = wave; t < kNT; t += kWAVES) {
        const int n0 = t * 16;
        const v4u* sp = (const v4u*)(sbase + (size_t)t * 32 * 32);
        union { v16bf bf; v4u u[2]; } ua, ub;
        ua.u[0] = sp[0]; ua.u[1] = sp[1];       // a0: 2x global_load_b128
        ub.u[0] = sp[2]; ub.u[1] = sp[3];       // a1
        v8f acc = {};
        acc = __builtin_amdgcn_wmma_f32_16x16x32_bf16(false, ua.bf, false, bf0,
                                                      (short)0, acc, false, false);
        acc = __builtin_amdgcn_wmma_f32_16x16x32_bf16(false, ub.bf, false, bf1,
                                                      (short)0, acc, false, false);
        #pragma unroll
        for (int v = 0; v < 8; ++v) {
            // x2 = log2(e)/8 * (2*dot - a - c)   (a, c pre-scaled)
            float x2 = acc[v] * kScale2 - anorm[n0 + v + 8 * hlf] - cc;
            float e  = __builtin_amdgcn_exp2f(x2);   // affinity <= 0: no max shift
            if (v & 1) psum1 += e; else psum0 += e;  // two chains, half the latency
        }
    }
    float psum = psum0 + psum1;
    // lanes l and l+16 hold the same column: fold halves, then fold 8 waves
    psum += __shfl_xor(psum, 16, 32);
    if (hlf == 0) redbuf[wave][lcol] = psum;
    __syncthreads();
    if (tid < 16) {
        float s = 0.f;
        #pragma unroll
        for (int w = 0; w < kWAVES; ++w) s += redbuf[w][tid];
        colrecip[tid] = 1.f / s;
    }
    __syncthreads();
    const float rs = colrecip[lcol];

    // ================= PASS 2: recompute, normalize, stream out =============
    float* ob = out + (size_t)b * kN * kM + m0;
    for (int t = wave; t < kNT; t += kWAVES) {
        const int n0 = t * 16;
        const v4u* sp = (const v4u*)(sbase + (size_t)t * 32 * 32);
        union { v16bf bf; v4u u[2]; } ua, ub;
        ua.u[0] = sp[0]; ua.u[1] = sp[1];
        ub.u[0] = sp[2]; ub.u[1] = sp[3];
        v8f acc = {};
        acc = __builtin_amdgcn_wmma_f32_16x16x32_bf16(false, ua.bf, false, bf0,
                                                      (short)0, acc, false, false);
        acc = __builtin_amdgcn_wmma_f32_16x16x32_bf16(false, ub.bf, false, bf1,
                                                      (short)0, acc, false, false);
        #pragma unroll
        for (int v = 0; v < 8; ++v) {
            const int n = n0 + v + 8 * hlf;
            float x2 = acc[v] * kScale2 - anorm[n] - cc;
            // write-once 260 MB stream: non-temporal, keep it out of L2
            __builtin_nontemporal_store(__builtin_amdgcn_exp2f(x2) * rs,
                                        &ob[(size_t)n * kM + lcol]);
        }
    }
}

// ---------------------------------------------------------------------------
// Fallback (no workspace dependency): self-contained version.
// ---------------------------------------------------------------------------
__global__ __launch_bounds__(256)
void affinity_softmax_selfcontained(const float* __restrict__ mk,
                                    const float* __restrict__ qk,
                                    float* __restrict__ out)
{
    const int b    = blockIdx.y;
    const int m0   = blockIdx.x * 16;
    const int tid  = threadIdx.x;
    const int lane = tid & 31;
    const int wave = tid >> 5;
    const int hlf  = lane >> 4;
    const int lcol = lane & 15;

    const float* mkb = mk + (size_t)b * kCK * kN;
    const float* qkb = qk + (size_t)b * kCK * kM;

    __shared__ float anorm[kN];                 // pre-scaled by kScale
    __shared__ float cterm[16];
    __shared__ float redbuf[kWAVES][16];
    __shared__ float colrecip[16];

    if (tid < 16) {
        float s = 0.f;
        const float* q = qkb + m0 + tid;
        for (int k = 0; k < kCK; ++k) { float v = q[(size_t)k * kM]; s += v * v; }
        cterm[tid] = s * kScale;
    }
    for (int n = tid; n < kN; n += 256) {
        float s = 0.f;
        const float* p = mkb + n;
        #pragma unroll 8
        for (int k = 0; k < kCK; ++k) { float v = p[(size_t)k * kN]; s += v * v; }
        anorm[n] = s * kScale;
    }

    v16bf bf0, bf1;
    {
        const float* q = qkb + m0 + lcol;
        #pragma unroll
        for (int e = 0; e < 16; ++e) {
            int k = kb_idx(e, hlf);
            bf0[e] = f2bf(q[(size_t)k * kM]);
            bf1[e] = f2bf(q[(size_t)(k + 32) * kM]);
        }
    }
    __syncthreads();
    const float cc = cterm[lcol];

    float psum0 = 0.f, psum1 = 0.f;
    for (int t = wave; t < kNT; t += kWAVES) {
        const int n0 = t * 16;
        v16bf a0, a1;
        const float* p = mkb + n0 + lcol;
        #pragma unroll
        for (int e = 0; e < 16; ++e) {
            int k = ka_idx(e, hlf);
            a0[e] = f2bf(p[(size_t)k * kN]);
            a1[e] = f2bf(p[(size_t)(k + 32) * kN]);
        }
        v8f acc = {};
        acc = __builtin_amdgcn_wmma_f32_16x16x32_bf16(false, a0, false, bf0,
                                                      (short)0, acc, false, false);
        acc = __builtin_amdgcn_wmma_f32_16x16x32_bf16(false, a1, false, bf1,
                                                      (short)0, acc, false, false);
        #pragma unroll
        for (int v = 0; v < 8; ++v) {
            float x2 = acc[v] * kScale2 - anorm[n0 + v + 8 * hlf] - cc;
            float e  = __builtin_amdgcn_exp2f(x2);
            if (v & 1) psum1 += e; else psum0 += e;
        }
    }
    float psum = psum0 + psum1;
    psum += __shfl_xor(psum, 16, 32);
    if (hlf == 0) redbuf[wave][lcol] = psum;
    __syncthreads();
    if (tid < 16) {
        float s = 0.f;
        #pragma unroll
        for (int w = 0; w < kWAVES; ++w) s += redbuf[w][tid];
        colrecip[tid] = 1.f / s;
    }
    __syncthreads();
    const float rs = colrecip[lcol];

    float* ob = out + (size_t)b * kN * kM + m0;
    for (int t = wave; t < kNT; t += kWAVES) {
        const int n0 = t * 16;
        v16bf a0, a1;
        const float* p = mkb + n0 + lcol;
        #pragma unroll
        for (int e = 0; e < 16; ++e) {
            int k = ka_idx(e, hlf);
            a0[e] = f2bf(p[(size_t)k * kN]);
            a1[e] = f2bf(p[(size_t)(k + 32) * kN]);
        }
        v8f acc = {};
        acc = __builtin_amdgcn_wmma_f32_16x16x32_bf16(false, a0, false, bf0,
                                                      (short)0, acc, false, false);
        acc = __builtin_amdgcn_wmma_f32_16x16x32_bf16(false, a1, false, bf1,
                                                      (short)0, acc, false, false);
        #pragma unroll
        for (int v = 0; v < 8; ++v) {
            const int n = n0 + v + 8 * hlf;
            float x2 = acc[v] * kScale2 - anorm[n] - cc;
            __builtin_nontemporal_store(__builtin_amdgcn_exp2f(x2) * rs,
                                        &ob[(size_t)n * kM + lcol]);
        }
    }
}

extern "C" void kernel_launch(void* const* d_in, const int* in_sizes, int n_in,
                              void* d_out, int out_size, void* d_ws, size_t ws_size,
                              hipStream_t stream) {
    const float* mk = (const float*)d_in[0];   // [B, CK, 1, H, W] f32
    const float* qk = (const float*)d_in[1];   // [B, CK, H, W]    f32
    float* out = (float*)d_out;                // [B, N, M] f32

    dim3 grid(kM / 16, kB);                    // 252 x 4 workgroups

    if (ws_size >= kWsNeeded && d_ws != nullptr) {
        char* ws = (char*)d_ws;
        __bf16* stage = (__bf16*)ws;
        float*  anorm = (float*)(ws + kAnormOff);
        float*  cnorm = (float*)(ws + kCnormOff);
        prepass_kernel<<<dim3(kNT, kB), 32, 0, stream>>>(mk, qk, stage, anorm, cnorm);
        affinity_softmax_fast<<<grid, 256, 0, stream>>>(qk, stage, anorm, cnorm, out);
    } else {
        affinity_softmax_selfcontained<<<grid, 256, 0, stream>>>(mk, qk, out);
    }
}
